// MobilityPyGEncoder_53532472377745
// MI455X (gfx1250) — matrix operand order repaired and checked by reading, not compiled, not run on hardware
//
#include <hip/hip_runtime.h>

typedef __attribute__((ext_vector_type(16))) _Float16 v16h;
typedef __attribute__((ext_vector_type(8)))  float    v8f;

// ---------------- degree / normalization ----------------

__global__ void k_fill_deg(float* __restrict__ deg, int n) {
  int i = blockIdx.x * blockDim.x + threadIdx.x;
  if (i < n) deg[i] = 1.0f;  // self-loop weight 1
}

__global__ void k_deg_acc(const int* __restrict__ dst, const float* __restrict__ w,
                          float* __restrict__ deg, int E) {
  int e = blockIdx.x * blockDim.x + threadIdx.x;
  if (e < E) atomicAdd(&deg[dst[e]], w[e]);
}

__global__ void k_dinv(const float* __restrict__ deg, float* __restrict__ dinv, int n) {
  int i = blockIdx.x * blockDim.x + threadIdx.x;
  if (i < n) {
    float d = deg[i];
    dinv[i] = (d > 0.f) ? rsqrtf(d) : 0.f;
  }
}

// ---------------- weight convert: W[k][n] fp32 -> Wt[n][k] f16 ----------------

__global__ void k_convw(const float* __restrict__ W1, const float* __restrict__ W2,
                        _Float16* __restrict__ Wt1, _Float16* __restrict__ Wt2) {
  int i = blockIdx.x * blockDim.x + threadIdx.x;
  if (i >= 2 * 128 * 128) return;
  const float* W = (i < 128 * 128) ? W1 : W2;
  _Float16*   Wt = (i < 128 * 128) ? Wt1 : Wt2;
  int j = i & (128 * 128 - 1);
  int k = j >> 7, n = j & 127;
  Wt[n * 128 + k] = (_Float16)W[k * 128 + n];
}

// ---------------- WMMA GEMM: Y[nrows,128] = (relu?)X[nrows,128] @ W ----------------
// One wave per 16-row tile; 8 column tiles; K loop in steps of 32.

__global__ void __launch_bounds__(256)
k_gemm_wmma(const float* __restrict__ X, const _Float16* __restrict__ Wt,
            float* __restrict__ Y, int nrows, int do_relu) {
  const int wave    = (int)((blockIdx.x * blockDim.x + threadIdx.x) >> 5);
  const int lane    = threadIdx.x & 31;
  const int rowbase = wave * 16;
  if (rowbase >= nrows) return;          // wave-uniform: EXEC stays all-ones
  const int lm   = lane & 15;
  const int half = lane >> 4;

  int arow = rowbase + lm;
  if (arow >= nrows) arow = nrows - 1;   // clamp: safe read, result row not stored
  const float* xrow = X + (size_t)arow * 128;

  v8f acc[8];
  const v8f zero = {0.f, 0.f, 0.f, 0.f, 0.f, 0.f, 0.f, 0.f};
#pragma unroll
  for (int t = 0; t < 8; ++t) acc[t] = zero;

#pragma unroll
  for (int kb = 0; kb < 128; kb += 32) {
    const int k0 = kb + half * 16;
    float4 a0 = *(const float4*)(xrow + k0);
    float4 a1 = *(const float4*)(xrow + k0 + 4);
    float4 a2 = *(const float4*)(xrow + k0 + 8);
    float4 a3 = *(const float4*)(xrow + k0 + 12);
    if (do_relu) {
      a0.x = fmaxf(a0.x, 0.f); a0.y = fmaxf(a0.y, 0.f);
      a0.z = fmaxf(a0.z, 0.f); a0.w = fmaxf(a0.w, 0.f);
      a1.x = fmaxf(a1.x, 0.f); a1.y = fmaxf(a1.y, 0.f);
      a1.z = fmaxf(a1.z, 0.f); a1.w = fmaxf(a1.w, 0.f);
      a2.x = fmaxf(a2.x, 0.f); a2.y = fmaxf(a2.y, 0.f);
      a2.z = fmaxf(a2.z, 0.f); a2.w = fmaxf(a2.w, 0.f);
      a3.x = fmaxf(a3.x, 0.f); a3.y = fmaxf(a3.y, 0.f);
      a3.z = fmaxf(a3.z, 0.f); a3.w = fmaxf(a3.w, 0.f);
    }
    // A fragment, contiguous K order (k = 16*half + s), same as B:
    // elements 0..7  -> K = kb + half*16 + 0..7
    // elements 8..15 -> K = kb + half*16 + 8..15
    v16h a;
    a[0]  = (_Float16)a0.x; a[1]  = (_Float16)a0.y;
    a[2]  = (_Float16)a0.z; a[3]  = (_Float16)a0.w;
    a[4]  = (_Float16)a1.x; a[5]  = (_Float16)a1.y;
    a[6]  = (_Float16)a1.z; a[7]  = (_Float16)a1.w;
    a[8]  = (_Float16)a2.x; a[9]  = (_Float16)a2.y;
    a[10] = (_Float16)a2.z; a[11] = (_Float16)a2.w;
    a[12] = (_Float16)a3.x; a[13] = (_Float16)a3.y;
    a[14] = (_Float16)a3.z; a[15] = (_Float16)a3.w;

    const int bko = kb + half * 16;  // B: lanes 0-15 hold K=kb..kb+15, lanes 16-31 next 16
#pragma unroll
    for (int ct = 0; ct < 8; ++ct) {
      v16h b = *(const v16h*)(Wt + (size_t)((ct * 16 + lm) * 128 + bko));
      acc[ct] = __builtin_amdgcn_wmma_f32_16x16x32_f16(
          false, a, false, b, (short)0, acc[ct], false, false);
    }
  }

  // C/D layout: VGPR j -> lanes 0-15: M=j, lanes 16-31: M=j+8; N = lane%16
#pragma unroll
  for (int ct = 0; ct < 8; ++ct) {
    const int col = ct * 16 + lm;
#pragma unroll
    for (int j = 0; j < 8; ++j) {
      const int r = rowbase + half * 8 + j;
      if (r < nrows) Y[(size_t)r * 128 + col] = acc[ct][j];
    }
  }
}

// ---------------- bias + self-loop init: Out = b + dinv^2 * Xw ----------------

__global__ void k_selfinit(const float* __restrict__ Xw, const float* __restrict__ dinv,
                           const float* __restrict__ bias, float* __restrict__ Out,
                           long total) {
  long idx = (long)blockIdx.x * blockDim.x + threadIdx.x;
  if (idx >= total) return;
  int node = (int)(idx >> 7);
  int c    = (int)(idx & 127);
  float di = dinv[node];
  Out[idx] = bias[c] + di * di * Xw[idx];
}

// ---------------- edge scatter: one wave per edge, float4 per lane ----------------

__global__ void k_scatter(const int* __restrict__ src, const int* __restrict__ dst,
                          const float* __restrict__ w, const float* __restrict__ dinv,
                          const float* __restrict__ Xw, float* __restrict__ Out, int E) {
  long gid = (long)blockIdx.x * blockDim.x + threadIdx.x;
  int e    = (int)(gid >> 5);
  int lane = (int)(gid & 31);
  if (e >= E) return;
  int s = src[e], d = dst[e];
  float nrm = dinv[s] * w[e] * dinv[d];
  const float4 v = *(const float4*)(Xw + (size_t)s * 128 + lane * 4);
  float* o = Out + (size_t)d * 128 + lane * 4;
  atomicAdd(o + 0, nrm * v.x);
  atomicAdd(o + 1, nrm * v.y);
  atomicAdd(o + 2, nrm * v.z);
  atomicAdd(o + 3, nrm * v.w);
}

// ---------------- launcher ----------------

extern "C" void kernel_launch(void* const* d_in, const int* in_sizes, int n_in,
                              void* d_out, int out_size, void* d_ws, size_t ws_size,
                              hipStream_t stream) {
  const float* x  = (const float*)d_in[0];
  const int*   ei = (const int*)d_in[1];   // [2, E]: row 0 = src, row 1 = dst
  const float* ew = (const float*)d_in[2];
  const float* W1 = (const float*)d_in[3];
  const float* b1 = (const float*)d_in[4];
  const float* W2 = (const float*)d_in[5];
  const float* b2 = (const float*)d_in[6];
  float* out = (float*)d_out;

  const int E = in_sizes[2];
  const int N = in_sizes[0] / 128;
  const int* src = ei;
  const int* dst = ei + E;

  // workspace layout
  float* deg  = (float*)d_ws;                       // N
  float* dinv = deg + N;                            // N
  float* xw   = dinv + N;                           // N*128 (reused for layer 2)
  float* h    = xw + (size_t)N * 128;               // N*128
  _Float16* Wt1 = (_Float16*)(h + (size_t)N * 128); // 128*128 f16
  _Float16* Wt2 = Wt1 + 128 * 128;                  // 128*128 f16

  const int TB = 256;
  const int gN     = (N + TB - 1) / TB;
  const int gE     = (E + TB - 1) / TB;
  const int gElem  = (int)(((long)N * 128 + TB - 1) / TB);
  const int gEdgeW = (int)(((long)E * 32 + TB - 1) / TB);
  const int waves  = (N + 15) / 16;
  const int gGemm  = (waves + 7) / 8;               // 8 waves (256 threads) per block
  const int gConv  = (2 * 128 * 128 + TB - 1) / TB;

  // normalization
  k_fill_deg<<<gN, TB, 0, stream>>>(deg, N);
  k_deg_acc<<<gE, TB, 0, stream>>>(dst, ew, deg, E);
  k_dinv<<<gN, TB, 0, stream>>>(deg, dinv, N);

  // weights -> f16 transposed
  k_convw<<<gConv, TB, 0, stream>>>(W1, W2, Wt1, Wt2);

  // layer 1
  k_gemm_wmma<<<gGemm, TB, 0, stream>>>(x, Wt1, xw, N, /*relu=*/0);
  k_selfinit<<<gElem, TB, 0, stream>>>(xw, dinv, b1, h, (long)N * 128);
  k_scatter<<<gEdgeW, TB, 0, stream>>>(src, dst, ew, dinv, xw, h, E);

  // layer 2 (ReLU folded into GEMM A-load; xw buffer reused)
  k_gemm_wmma<<<gGemm, TB, 0, stream>>>(h, Wt2, xw, N, /*relu=*/1);
  k_selfinit<<<gElem, TB, 0, stream>>>(xw, dinv, b2, out, (long)N * 128);
  k_scatter<<<gEdgeW, TB, 0, stream>>>(src, dst, ew, dinv, xw, out, E);
}
